// ViT_44160853738019
// MI455X (gfx1250) — matrix operand order repaired and checked by reading, not compiled, not run on hardware
//
#include <hip/hip_runtime.h>
#include <math.h>

// ---------------- types ----------------
typedef __attribute__((ext_vector_type(16))) _Float16 v16h;
typedef __attribute__((ext_vector_type(8)))  _Float16 v8h;
typedef __attribute__((ext_vector_type(8)))  float    v8f;
typedef __attribute__((ext_vector_type(4)))  unsigned int v4u;
typedef __attribute__((ext_vector_type(8)))  int      v8i;
typedef __attribute__((ext_vector_type(4)))  int      v4i;

union FragU { v16h v; v8h h[2]; };

// ---------------- constants ----------------
#define BB 4
#define CC 128
#define HH 512
#define PP 16
#define GH 32
#define GW 32
#define NTOK 1024
#define PATCH_DIM 32768
#define DIM 512
#define MLP 256
#define DHD 64

// TDM (Tensor Data Mover) availability
#if defined(__has_builtin)
#if __has_builtin(__builtin_amdgcn_tensor_load_to_lds) && __has_builtin(__builtin_amdgcn_s_wait_tensorcnt)
#define USE_TDM 1
#endif
#endif
#ifndef USE_TDM
#define USE_TDM 0
#endif

__device__ __forceinline__ float gelu_exact(float v) {
    return 0.5f * v * (1.0f + erff(v * 0.70710678118654752f));
}

#if USE_TDM
// Issue one TDM 2D tile load: rows x 32 f16 elements, row stride = strideElems.
// Rows >= validRows read OOB -> zero-filled in LDS (D#.tensor_dim1 bound).
// D# bit layout per CDNA5 ISA (group0 128b, group1 256b).
// Toolchain uses the 6-arg builtin: (v4u g0, v8i g1, v4i g2, v4i g3, v8i g4, i32 cpol).
__device__ __forceinline__ void tdm_load_tile(unsigned ldsOff, const _Float16* gptr,
                                              int tileRows, int validRows,
                                              int strideElems) {
    const unsigned long long ga = (unsigned long long)(uintptr_t)gptr;
    v4u g0;
    g0[0] = 1u;                                              // count=1, user mode
    g0[1] = ldsOff;                                          // lds_addr
    g0[2] = (unsigned)(ga & 0xFFFFFFFFu);                    // global_addr[31:0]
    g0[3] = (unsigned)((ga >> 32) & 0x01FFFFFFu) | (2u << 30); // global_addr[56:32] | type=2
    v8i g1;
    g1[0] = (int)(1u << 16);                                 // data_size=1 (2 bytes), mask=0
    g1[1] = (int)(32u << 16);                                // tensor_dim0 lo16 = 32
    g1[2] = (int)((unsigned)validRows << 16);                // tensor_dim0 hi=0 | tensor_dim1 lo16
    g1[3] = (int)(((unsigned)validRows >> 16) | (32u << 16)); // tensor_dim1 hi | tile_dim0=32
    g1[4] = (int)(unsigned)tileRows;                         // tile_dim1 | tile_dim2=0
    g1[5] = (int)(unsigned)strideElems;                      // tensor_dim0_stride lo32
    g1[6] = 0;                                               // stride hi16 | dim1_stride lo16
    g1[7] = 0;
    v4i g2 = {0, 0, 0, 0};
    v4i g3 = {0, 0, 0, 0};
    v8i g4 = {0, 0, 0, 0, 0, 0, 0, 0};
    __builtin_amdgcn_tensor_load_to_lds(g0, g1, g2, g3, g4, 0);
}
#endif

// Shared-memory layout (single block, offsets are LDS byte addresses)
#define ASOFF0 0u
#define ASOFF1 4096u
#define BSOFF0 8192u
#define BSOFF1 16384u
#define SMEM_BYTES 24576

// One K=32 step: A tile 64x32 @As, Bt tile 128x32 @Bs; wave does 16x64 via 4 WMMA.
__device__ __forceinline__ void wmma_step(const _Float16* __restrict__ As,
                                          const _Float16* __restrict__ Bs,
                                          v8f acc[4], int wm, int wn, int lrow, int lhalf) {
    FragU fa;
    fa.h[0] = *(const v8h*)&As[(wm + lrow) * 32 + lhalf * 8];
    fa.h[1] = *(const v8h*)&As[(wm + lrow) * 32 + 16 + lhalf * 8];
    FragU fb[4];
#pragma unroll
    for (int j = 0; j < 4; ++j) {
        const int c = wn + j * 16 + lrow;
        fb[j].h[0] = *(const v8h*)&Bs[c * 32 + lhalf * 8];
        fb[j].h[1] = *(const v8h*)&Bs[c * 32 + 16 + lhalf * 8];
    }
#pragma unroll
    for (int j = 0; j < 4; ++j) {
        acc[j] = __builtin_amdgcn_wmma_f32_16x16x32_f16(
            false, fa.v, false, fb[j].v, (short)0, acc[j], false, false);
    }
}

// ---------------- img -> f16 patch matrix [B*N, PATCH_DIM], k=(p1*P+p2)*C+c ----
__global__ __launch_bounds__(256) void k_patches(const float* __restrict__ img,
                                                 _Float16* __restrict__ out) {
    long long idx = (long long)blockIdx.x * 256 + threadIdx.x;   // B*N*PATCH_DIM total
    const int k  = (int)(idx & (PATCH_DIM - 1));
    const long long bn = idx >> 15;
    const int n  = (int)(bn & (NTOK - 1));
    const int b  = (int)(bn >> 10);
    const int c  = k & (CC - 1);
    const int pp = k >> 7;            // p1*P + p2
    const int p2 = pp & (PP - 1);
    const int p1 = pp >> 4;
    const int gh = n >> 5, gw = n & 31;
    const long long src = (((long long)(b * CC + c) * HH) + gh * PP + p1) * HH + gw * PP + p2;
    out[idx] = (_Float16)img[src];
}

// ---------------- w[K,N] fp32 -> wT[N,K] f16 ----------------
__global__ __launch_bounds__(256) void k_transpose(const float* __restrict__ w,
                                                   _Float16* __restrict__ wT,
                                                   int rows, int cols, long long total) {
    long long idx = (long long)blockIdx.x * 256 + threadIdx.x;
    if (idx >= total) return;
    const int n = (int)(idx / rows);
    const int k = (int)(idx % rows);
    wT[idx] = (_Float16)w[(size_t)k * cols + n];
}

// ---------------- LayerNorm over DIM=512, one row per block, f16 out ----------
__global__ __launch_bounds__(256) void k_ln(const float* __restrict__ x,
                                            const float* __restrict__ s,
                                            const float* __restrict__ b,
                                            _Float16* __restrict__ out) {
    __shared__ float red[256];
    const int row = blockIdx.x, t = threadIdx.x;
    const size_t base = (size_t)row * DIM;
    const float v0 = x[base + t];
    const float v1 = x[base + 256 + t];
    red[t] = v0 + v1;
    __syncthreads();
    for (int o = 128; o > 0; o >>= 1) { if (t < o) red[t] += red[t + o]; __syncthreads(); }
    const float mean = red[0] * (1.0f / DIM);
    __syncthreads();
    const float d0 = v0 - mean, d1 = v1 - mean;
    red[t] = d0 * d0 + d1 * d1;
    __syncthreads();
    for (int o = 128; o > 0; o >>= 1) { if (t < o) red[t] += red[t + o]; __syncthreads(); }
    const float rstd = rsqrtf(red[0] * (1.0f / DIM) + 1e-5f);
    out[base + t]       = (_Float16)(d0 * rstd * s[t]       + b[t]);
    out[base + 256 + t] = (_Float16)(d1 * rstd * s[256 + t] + b[256 + t]);
}

// ---------------- WMMA GEMM: C[M,N] = A[M,K] * Bt[N,K]^T (+epilogues) --------
// 256 threads = 8 waves; block tile 64x128; wave tile 16x64; K step 32.
// TDM path: wave0 DMAs A/B tiles to double-buffered LDS, s_wait_tensorcnt pipeline.
template <int OUTF16, int USE_BIAS, int USE_RES, int USE_POS, int USE_GELU>
__global__ __launch_bounds__(256) void k_gemm(const _Float16* __restrict__ A,
                                              const _Float16* __restrict__ Bt,
                                              float* __restrict__ Cf,
                                              _Float16* __restrict__ Ch,
                                              const float* __restrict__ bias,
                                              const float* __restrict__ res,
                                              const float* __restrict__ pos,
                                              int M, int N, int K,
                                              long long aZ, long long bZ, long long cZ) {
    __shared__ __align__(16) char smem[SMEM_BYTES];
    (void)M;

    const int t = threadIdx.x;
    const int z = blockIdx.z;
    A  += (size_t)z * aZ;
    Bt += (size_t)z * bZ;
    const size_t czoff = (size_t)z * cZ;

    const int blockRow = blockIdx.y * 64;
    const int blockCol = blockIdx.x * 128;
    const int lane = t & 31, w = t >> 5;
    const int wm = (w & 3) * 16;   // M sub-tile
    const int wn = (w >> 2) * 64;  // N sub-tile
    const int lrow = lane & 15, lhalf = lane >> 4;

    v8f acc[4] = {};

#if USE_TDM
    const int nsteps = K >> 5;
    const int validB = (N - blockCol) < 128 ? (N - blockCol) : 128;
    const _Float16* Abase = A + (size_t)blockRow * K;
    const _Float16* Bbase = Bt + (size_t)blockCol * K;
    if (w == 0) {                                   // wave-uniform branch: one issuer
        tdm_load_tile(ASOFF0, Abase, 64, 64, K);
        tdm_load_tile(BSOFF0, Bbase, 128, validB, K);
    }
    for (int s = 0; s < nsteps; ++s) {
        const int cur = s & 1;
        if (w == 0) {
            if (s + 1 < nsteps) {
                const int kk = (s + 1) << 5;
                tdm_load_tile(cur ? ASOFF0 : ASOFF1, Abase + kk, 64, 64, K);
                tdm_load_tile(cur ? BSOFF0 : BSOFF1, Bbase + kk, 128, validB, K);
                __builtin_amdgcn_s_wait_tensorcnt(2);   // tile s complete (in-order)
            } else {
                __builtin_amdgcn_s_wait_tensorcnt(0);
            }
        }
        __syncthreads();
        wmma_step((const _Float16*)(smem + (cur ? ASOFF1 : ASOFF0)),
                  (const _Float16*)(smem + (cur ? BSOFF1 : BSOFF0)),
                  acc, wm, wn, lrow, lhalf);
        __syncthreads();
    }
#else
    _Float16* As = (_Float16*)(smem + ASOFF0);
    _Float16* Bs = (_Float16*)(smem + BSOFF0);
    const int arow = t >> 2, acol = (t & 3) * 8;   // A stage: 64 rows x 32, 16B/thread
    const int brow = t >> 1, bcol = (t & 1) * 16;  // B stage: 128 rows x 32, 32B/thread
    const int bN = blockCol + brow;
    const int bNc = bN < (N - 1) ? bN : (N - 1);   // clamp: junk cols are never stored
    for (int kk = 0; kk < K; kk += 32) {
        {
            const _Float16* src = A + (size_t)(blockRow + arow) * K + kk + acol;
            *(uint4*)&As[arow * 32 + acol] = *(const uint4*)src;
            if (kk + 32 < K) __builtin_prefetch((const void*)(src + 32), 0, 1);
        }
        {
            const _Float16* src = Bt + (size_t)bNc * K + kk + bcol;
            *(uint4*)&Bs[brow * 32 + bcol]     = *(const uint4*)src;
            *(uint4*)&Bs[brow * 32 + bcol + 8] = *(const uint4*)(src + 8);
            if (kk + 32 < K) __builtin_prefetch((const void*)(src + 32), 0, 1);
        }
        __syncthreads();
        wmma_step(As, Bs, acc, wm, wn, lrow, lhalf);
        __syncthreads();
    }
#endif

#pragma unroll
    for (int j = 0; j < 4; ++j) {
        const int g = blockCol + wn + j * 16 + lrow;
        if (g < N) {
#pragma unroll
            for (int i = 0; i < 8; ++i) {
                const int row = blockRow + wm + lhalf * 8 + i;
                float v = acc[j][i];
                if (USE_BIAS) v += bias[g];
                if (USE_POS)  v += pos[(size_t)(row & (NTOK - 1)) * N + g];
                if (USE_RES)  v += res[czoff + (size_t)row * N + g];
                if (USE_GELU) v = gelu_exact(v);
                if (OUTF16) Ch[czoff + (size_t)row * N + g] = (_Float16)v;
                else        Cf[czoff + (size_t)row * N + g] = v;
            }
        }
    }
}

// ---------------- qkv fp32 [B*N, 3*inner] -> q,k f16 [B,h,N,DH], vT [B,h,DH,N]
__global__ __launch_bounds__(256) void k_qkv_split(const float* __restrict__ qkv,
                                                   _Float16* __restrict__ q,
                                                   _Float16* __restrict__ k,
                                                   _Float16* __restrict__ vT,
                                                   int heads, long long total) {
    long long idx = (long long)blockIdx.x * 256 + threadIdx.x;
    if (idx >= total) return;
    const int d  = (int)(idx & (DHD - 1));
    long long r  = idx >> 6;
    const int hh = (int)(r % heads);  r /= heads;
    const int n  = (int)(r & (NTOK - 1));
    const int b  = (int)(r >> 10);
    const int inner = heads * DHD;
    const size_t sb = ((size_t)(b * NTOK + n)) * 3 * inner + hh * DHD + d;
    const size_t qi = ((size_t)(b * heads + hh) * NTOK + n) * DHD + d;
    q[qi] = (_Float16)(qkv[sb] * 0.125f);            // fold DH^-0.5
    k[qi] = (_Float16)qkv[sb + inner];
    vT[((size_t)(b * heads + hh) * DHD + d) * NTOK + n] = (_Float16)qkv[sb + 2 * inner];
}

// ---------------- softmax(scores + rel_bias) per row, per image b ------------
__global__ __launch_bounds__(256) void k_softmax(const float* __restrict__ scores,
                                                 const float* __restrict__ table,
                                                 _Float16* __restrict__ attn,
                                                 int heads) {
    __shared__ float buf[NTOK];
    __shared__ float red[256];
    const int i = blockIdx.x, h = blockIdx.y, t = threadIdx.x;
    const size_t base = ((size_t)h * NTOK + i) * NTOK;
    const int gih = i >> 5, giw = i & 31;
    float mx = -1e30f;
#pragma unroll
    for (int kq = 0; kq < 4; ++kq) {
        const int j = t + kq * 256;
        const int gjh = j >> 5, gjw = j & 31;
        const int ridx = (gih - gjh + GH - 1) * (2 * GH - 1) + (giw - gjw + GW - 1);
        const float v = scores[base + j] + table[(size_t)ridx * heads + h];
        buf[j] = v;
        mx = fmaxf(mx, v);
    }
    red[t] = mx; __syncthreads();
    for (int o = 128; o > 0; o >>= 1) { if (t < o) red[t] = fmaxf(red[t], red[t + o]); __syncthreads(); }
    mx = red[0]; __syncthreads();
    float sum = 0.f;
#pragma unroll
    for (int kq = 0; kq < 4; ++kq) {
        const int j = t + kq * 256;
        const float e = __expf(buf[j] - mx);
        buf[j] = e;
        sum += e;
    }
    red[t] = sum; __syncthreads();
    for (int o = 128; o > 0; o >>= 1) { if (t < o) red[t] += red[t + o]; __syncthreads(); }
    const float inv = 1.0f / red[0];
#pragma unroll
    for (int kq = 0; kq < 4; ++kq) {
        const int j = t + kq * 256;
        attn[base + j] = (_Float16)(buf[j] * inv);
    }
}

// ---------------- o [B,h,N,DH] f16 -> o2 [B*N, h*DH] f16 ---------------------
__global__ __launch_bounds__(256) void k_permute_o(const _Float16* __restrict__ o,
                                                   _Float16* __restrict__ o2,
                                                   int heads, long long total) {
    long long idx = (long long)blockIdx.x * 256 + threadIdx.x;
    if (idx >= total) return;
    const int d  = (int)(idx & (DHD - 1));
    long long r  = idx >> 6;
    const int hh = (int)(r % heads);  r /= heads;
    const int n  = (int)(r & (NTOK - 1));
    const int b  = (int)(r >> 10);
    o2[((size_t)(b * NTOK + n)) * (heads * DHD) + hh * DHD + d] =
        o[((size_t)(b * heads + hh) * NTOK + n) * DHD + d];
}

// ---------------- x [B,N,DIM] -> out [B,DIM,GH,GW] fp32 ----------------------
__global__ __launch_bounds__(256) void k_final(const float* __restrict__ x,
                                               float* __restrict__ out) {
    long long idx = (long long)blockIdx.x * 256 + threadIdx.x;  // B*DIM*N
    const int gw = (int)(idx & 31);
    const int gh = (int)((idx >> 5) & 31);
    const int d  = (int)((idx >> 10) & (DIM - 1));
    const int b  = (int)(idx >> 19);
    out[idx] = x[((size_t)(b * NTOK) + gh * GW + gw) * DIM + d];
}

// =============================================================================
extern "C" void kernel_launch(void* const* d_in, const int* in_sizes, int n_in,
                              void* d_out, int out_size, void* d_ws, size_t ws_size,
                              hipStream_t stream) {
    (void)in_sizes; (void)n_in; (void)out_size; (void)ws_size;

    const float* img     = (const float*)d_in[0];
    const float* patch_w = (const float*)d_in[1];
    const float* patch_b = (const float*)d_in[2];
    const float* pos_emb = (const float*)d_in[3];
    const float* ln1_s   = (const float*)d_in[4];
    const float* ln1_b   = (const float*)d_in[5];
    const float* ln2_s   = (const float*)d_in[6];
    const float* ln2_b   = (const float*)d_in[7];
    const float* qkv_w[3] = {(const float*)d_in[8],  (const float*)d_in[12], (const float*)d_in[16]};
    const float* out_w[3] = {(const float*)d_in[9],  (const float*)d_in[13], (const float*)d_in[17]};
    const float* out_b[3] = {(const float*)d_in[10], (const float*)d_in[14], (const float*)d_in[18]};
    const float* btab[3]  = {(const float*)d_in[11], (const float*)d_in[15], (const float*)d_in[19]};
    const float* ff_w1 = (const float*)d_in[20];
    const float* ff_b1 = (const float*)d_in[21];
    const float* ff_w2 = (const float*)d_in[22];
    const float* ff_b2 = (const float*)d_in[23];

    const int heads_arr[3] = {2, 4, 8};
    const long long ROWS = (long long)BB * NTOK;   // 4096

    // ---- workspace carve (256B aligned slabs) ----
    char* wsp = (char*)d_ws;
    #define WS_ALLOC(var, T, bytes) T* var = (T*)wsp; wsp += (((size_t)(bytes)) + 255) & ~(size_t)255;
    WS_ALLOC(patches,  _Float16, ROWS * PATCH_DIM * 2)              // 256 MB
    WS_ALLOC(patch_wT, _Float16, (size_t)DIM * PATCH_DIM * 2)       // 32 MB
    WS_ALLOC(x,        float,    ROWS * DIM * 4)                    // 8 MB fp32 residual stream
    WS_ALLOC(xn,       _Float16, ROWS * DIM * 2)                    // 4 MB
    WS_ALLOC(wbuf,     _Float16, (size_t)1536 * DIM * 2)            // 1.5 MB reusable f16 weights
    WS_ALLOC(qkvbuf,   float,    ROWS * 1536 * 4)                   // 24 MB
    WS_ALLOC(qbuf,     _Float16, (size_t)BB * 8 * NTOK * DHD * 2)   // 4 MB
    WS_ALLOC(kbuf,     _Float16, (size_t)BB * 8 * NTOK * DHD * 2)   // 4 MB
    WS_ALLOC(vTbuf,    _Float16, (size_t)BB * 8 * DHD * NTOK * 2)   // 4 MB
    WS_ALLOC(scoresB,  float,    (size_t)8 * NTOK * NTOK * 4)       // 32 MB (per image b)
    WS_ALLOC(attnB,    _Float16, (size_t)8 * NTOK * NTOK * 2)       // 16 MB (per image b)
    WS_ALLOC(obuf,     _Float16, (size_t)BB * 8 * NTOK * DHD * 2)   // 4 MB
    WS_ALLOC(o2buf,    _Float16, ROWS * DIM * 2)                    // 4 MB
    WS_ALLOC(hmid,     _Float16, ROWS * MLP * 2)                    // 2 MB
    #undef WS_ALLOC

    // ---- patch embed ----
    {
        long long tot = ROWS * PATCH_DIM;
        k_patches<<<dim3((unsigned)(tot / 256)), 256, 0, stream>>>(img, patches);
        long long wt = (long long)DIM * PATCH_DIM;
        k_transpose<<<dim3((unsigned)((wt + 255) / 256)), 256, 0, stream>>>(
            patch_w, patch_wT, PATCH_DIM, DIM, wt);
        // x = patches @ patch_w + patch_b + pos_emb
        k_gemm<0, 1, 0, 1, 0><<<dim3(DIM / 128, (unsigned)(ROWS / 64), 1), 256, 0, stream>>>(
            patches, patch_wT, x, nullptr, patch_b, nullptr, pos_emb,
            (int)ROWS, DIM, PATCH_DIM, 0, 0, 0);
    }

    for (int l = 0; l < 3; ++l) {
        const int heads = heads_arr[l];
        const int inner = heads * DHD;

        // ---- attention ----
        k_ln<<<dim3((unsigned)ROWS), 256, 0, stream>>>(x, ln1_s + l * DIM, ln1_b + l * DIM, xn);

        long long wt = (long long)3 * inner * DIM;
        k_transpose<<<dim3((unsigned)((wt + 255) / 256)), 256, 0, stream>>>(
            qkv_w[l], wbuf, DIM, 3 * inner, wt);
        k_gemm<0, 0, 0, 0, 0><<<dim3((3 * inner + 127) / 128, (unsigned)(ROWS / 64), 1), 256, 0, stream>>>(
            xn, wbuf, qkvbuf, nullptr, nullptr, nullptr, nullptr,
            (int)ROWS, 3 * inner, DIM, 0, 0, 0);

        long long spl = (long long)BB * NTOK * inner;
        k_qkv_split<<<dim3((unsigned)((spl + 255) / 256)), 256, 0, stream>>>(
            qkvbuf, qbuf, kbuf, vTbuf, heads, spl);

        for (int b = 0; b < BB; ++b) {
            const size_t boff = (size_t)b * heads * NTOK * DHD;
            // scores[h,N,N] = q @ k^T (scale folded into q)
            k_gemm<0, 0, 0, 0, 0><<<dim3(NTOK / 128, NTOK / 64, heads), 256, 0, stream>>>(
                qbuf + boff, kbuf + boff, scoresB, nullptr, nullptr, nullptr, nullptr,
                NTOK, NTOK, DHD, (long long)NTOK * DHD, (long long)NTOK * DHD,
                (long long)NTOK * NTOK);
            k_softmax<<<dim3(NTOK, heads), 256, 0, stream>>>(scoresB, btab[l], attnB, heads);
            // o[h,N,DH] = attn @ v   (Bt = vT)
            k_gemm<1, 0, 0, 0, 0><<<dim3(1, NTOK / 64, heads), 256, 0, stream>>>(
                attnB, vTbuf + boff, nullptr, obuf + boff, nullptr, nullptr, nullptr,
                NTOK, DHD, NTOK, (long long)NTOK * NTOK, (long long)DHD * NTOK,
                (long long)NTOK * DHD);
        }

        long long perm = (long long)BB * NTOK * inner;
        k_permute_o<<<dim3((unsigned)((perm + 255) / 256)), 256, 0, stream>>>(
            obuf, o2buf, heads, perm);

        wt = (long long)DIM * inner;
        k_transpose<<<dim3((unsigned)((wt + 255) / 256)), 256, 0, stream>>>(
            out_w[l], wbuf, inner, DIM, wt);
        // x = o2 @ out_w + out_b + x   (in-place residual: each thread RMWs its own element)
        k_gemm<0, 1, 1, 0, 0><<<dim3(DIM / 128, (unsigned)(ROWS / 64), 1), 256, 0, stream>>>(
            o2buf, wbuf, x, nullptr, out_b[l], x, nullptr, (int)ROWS, DIM, inner, 0, 0, 0);

        // ---- feed-forward ----
        k_ln<<<dim3((unsigned)ROWS), 256, 0, stream>>>(x, ln2_s + l * DIM, ln2_b + l * DIM, xn);

        wt = (long long)MLP * DIM;
        k_transpose<<<dim3((unsigned)((wt + 255) / 256)), 256, 0, stream>>>(
            ff_w1 + (size_t)l * DIM * MLP, wbuf, DIM, MLP, wt);
        k_gemm<1, 1, 0, 0, 1><<<dim3(MLP / 128, (unsigned)(ROWS / 64), 1), 256, 0, stream>>>(
            xn, wbuf, nullptr, hmid, ff_b1 + l * MLP, nullptr, nullptr,
            (int)ROWS, MLP, DIM, 0, 0, 0);

        wt = (long long)DIM * MLP;
        k_transpose<<<dim3((unsigned)((wt + 255) / 256)), 256, 0, stream>>>(
            ff_w2 + (size_t)l * MLP * DIM, wbuf, MLP, DIM, wt);
        k_gemm<0, 1, 1, 0, 0><<<dim3(DIM / 128, (unsigned)(ROWS / 64), 1), 256, 0, stream>>>(
            hmid, wbuf, x, nullptr, ff_b2 + l * DIM, x, nullptr,
            (int)ROWS, DIM, MLP, 0, 0, 0);
    }

    // ---- [B,N,DIM] -> [B,DIM,GH,GW] ----
    {
        long long tot = (long long)BB * DIM * NTOK;
        k_final<<<dim3((unsigned)(tot / 256)), 256, 0, stream>>>(x, (float*)d_out);
    }
}
